// SelfAttention_54915451847023
// MI455X (gfx1250) — matrix operand order repaired and checked by reading, not compiled, not run on hardware
//
#include <hip/hip_runtime.h>
#include <hip/hip_bf16.h>

// ---------------------------------------------------------------------------
// Self-attention forward for MI455X (gfx1250).
// bf16 WMMA + f32 accumulate, async global->LDS staging (ASYNCcnt path),
// double-buffered tiles, causal flash attention.
// B=2, S=2048, D=1024, H=16, DK=64.
// ---------------------------------------------------------------------------

typedef __bf16 bf16;
typedef __attribute__((ext_vector_type(8)))  __bf16 v8bf;
typedef __attribute__((ext_vector_type(16))) __bf16 v16bf;
typedef __attribute__((ext_vector_type(8)))  float  v8f;

#define ATTN_B  2
#define ATTN_S  2048
#define ATTN_D  1024
#define ATTN_H  16
#define ATTN_DK 64
#define ATTN_M  (ATTN_B * ATTN_S)   // 4096 rows for the projection GEMMs

// ----------------------------- helpers -------------------------------------

__device__ __forceinline__ unsigned short f2bfbits(float f) {
    union { float f; unsigned u; } v; v.f = f;
    unsigned u = v.u;
    unsigned r = u + 0x7FFFu + ((u >> 16) & 1u);   // round-to-nearest-even
    return (unsigned short)(r >> 16);
}

__device__ __forceinline__ v8f wmma_bf16(v16bf a, v16bf b, v8f c) {
    // (neg_a, A, neg_b, B, c_mod, C, reuse_a, reuse_b)
    return __builtin_amdgcn_wmma_f32_16x16x32_bf16(false, a, false, b,
                                                   (short)0, c, false, false);
}

// Async copy of one 16-byte chunk, global -> LDS (CDNA5 ASYNCcnt path).
// vdst = 32-bit LDS byte address, vaddr = 64-bit global address.
__device__ __forceinline__ void async_b128(const bf16* g, const bf16* l) {
    asm volatile("global_load_async_to_lds_b128 %0, %1, off"
                 :: "v"((unsigned)(size_t)l),
                    "v"((unsigned long long)(size_t)g)
                 : "memory");
}
__device__ __forceinline__ void wait_async_issued(int pending /*0 or 1 group*/,
                                                  int per_tile) {
    // in-order completion: waiting <= per_tile leaves only the newly issued
    // tile in flight; <= 0 drains everything.
    if (pending) {
        if (per_tile == 5) asm volatile("s_wait_asynccnt 0x5" ::: "memory");
        else               asm volatile("s_wait_asynccnt 0x4" ::: "memory");
    } else {
        asm volatile("s_wait_asynccnt 0x0" ::: "memory");
    }
}

// A-operand 16x32 bf16 tile, row-major source with leading dim `ld` (elems).
// Lane L<16: row L, K chunks [0..7],[16..23]; lane L>=16: row L-16,
// chunks [8..15],[24..31].  (ISA 7.12.2, 16-bit A layout.)
__device__ __forceinline__ v16bf load_a16x32(const bf16* base, int ld) {
    const int lane = threadIdx.x & 31;
    const int row  = lane & 15;
    const int half = lane >> 4;
    const bf16* p = base + (size_t)row * ld + half * 8;
    v8bf lo = *(const v8bf*)(p);
    v8bf hi = *(const v8bf*)(p + 16);
    return __builtin_shufflevector(lo, hi, 0,1,2,3,4,5,6,7,8,9,10,11,12,13,14,15);
}

// B-operand 32x16 bf16 tile from a K-contiguous matrix with leading dim `ld`:
// operand column n = memory row n; lane n holds k=0..7,16..23, lane n+16 k=8..15,24..31.
__device__ __forceinline__ v16bf load_b32x16(const bf16* base, int ld) {
    const int lane = threadIdx.x & 31;
    const int col  = lane & 15;
    const int half = lane >> 4;
    const bf16* p = base + (size_t)col * ld + half * 8;
    v8bf lo = *(const v8bf*)(p);
    v8bf hi = *(const v8bf*)(p + 16);
    return __builtin_shufflevector(lo, hi, 0,1,2,3,4,5,6,7,8,9,10,11,12,13,14,15);
}

// --------------------------- fp32 -> bf16 ----------------------------------

__global__ __launch_bounds__(256) void k_cvt_f32_bf16(const float* __restrict__ src,
                                                      unsigned short* __restrict__ dst,
                                                      int n) {
    int i = (blockIdx.x * 256 + threadIdx.x) * 4;
    if (i + 3 < n) {
        float4 f = *(const float4*)(src + i);
        ushort4 o;
        o.x = f2bfbits(f.x); o.y = f2bfbits(f.y);
        o.z = f2bfbits(f.z); o.w = f2bfbits(f.w);
        *(ushort4*)(dst + i) = o;
    }
}

// --------------------------- GEMM: C = A @ W^T -----------------------------
// A: (M,K) bf16 row-major.  W: (N,K) bf16 row-major (einsum 'mi,ni->mn').
// mode 0: bf16 out row-major (M,N)                       (Q, K projections)
// mode 1: bf16 out scattered to Vt (B,H,DK,S)            (V projection)
// mode 2: f32 out row-major (M,N)                        (final -> d_out)
// Block: 256 thr = 8 waves (4 M x 2 N).  Block tile 256x64, K-step 32.
// Double-buffered async staging: A tile 16KB + W tile 4KB per buffer.

__global__ __launch_bounds__(256) void k_gemm_xwt(const bf16* __restrict__ A,
                                                  const bf16* __restrict__ W,
                                                  void* __restrict__ out,
                                                  int M, int N, int K, int mode) {
    __shared__ __align__(32) bf16 ldsA[2][256 * 32];   // [buf][row*32 + k]
    __shared__ __align__(32) bf16 ldsW[2][64 * 32];    // [buf][row*32 + k]

    const int t    = threadIdx.x;
    const int lane = t & 31;
    const int wave = t >> 5;
    const int wm = wave & 3, wn = wave >> 2;
    const int m0 = blockIdx.x * 256;
    const int n0 = blockIdx.y * 64;
    const int col = lane & 15, half = lane >> 4;

    const int nk = K / 32;

    // Issue one tile's async copies: 4 A-chunks + 1 W-chunk per thread.
    auto issue_tile = [&](int kt, int buf) {
        const int k = kt * 32;
        // A tile: 256 rows x 32 elems; thread t owns row t (64B = 4 chunks).
        const bf16* ga = A + (size_t)(m0 + t) * K + k;
        bf16*       la = &ldsA[buf][t * 32];
        #pragma unroll
        for (int i = 0; i < 4; ++i)
            async_b128(ga + i * 8, la + i * 8);
        // W tile: 64 rows x 32 elems; 256 chunks, one per thread.
        const int wr = t >> 2, wsub = t & 3;
        async_b128(W + (size_t)(n0 + wr) * K + k + wsub * 8,
                   &ldsW[buf][wr * 32 + wsub * 8]);
    };

    v8f acc[4][2] = {};

    issue_tile(0, 0);
    for (int kt = 0; kt < nk; ++kt) {
        const int buf = kt & 1;
        const int more = (kt + 1 < nk);
        if (more) issue_tile(kt + 1, buf ^ 1);
        wait_async_issued(more, 5);
        __syncthreads();

        v16bf a[4], b[2];
        #pragma unroll
        for (int mt = 0; mt < 4; ++mt)
            a[mt] = load_a16x32(&ldsA[buf][(wm * 64 + 16 * mt) * 32], 32);
        #pragma unroll
        for (int nt = 0; nt < 2; ++nt)
            b[nt] = load_b32x16(&ldsW[buf][(wn * 32 + 16 * nt) * 32], 32);
        #pragma unroll
        for (int mt = 0; mt < 4; ++mt)
            #pragma unroll
            for (int nt = 0; nt < 2; ++nt)
                acc[mt][nt] = wmma_bf16(a[mt], b[nt], acc[mt][nt]);
        __syncthreads();
    }

    // C layout: VGPR r, lanes 0-15 -> (M=r, N=lane); lanes 16-31 -> (M=r+8).
    #pragma unroll
    for (int mt = 0; mt < 4; ++mt) {
        #pragma unroll
        for (int nt = 0; nt < 2; ++nt) {
            #pragma unroll
            for (int r = 0; r < 8; ++r) {
                const int m = m0 + wm * 64 + 16 * mt + r + 8 * half;
                const int n = n0 + wn * 32 + 16 * nt + col;
                const float v = acc[mt][nt][r];
                if (mode == 0) {
                    ((unsigned short*)out)[(size_t)m * N + n] = f2bfbits(v);
                } else if (mode == 1) {
                    const int b_ = m / ATTN_S, s_ = m % ATTN_S;
                    const int h_ = n / ATTN_DK, dk = n % ATTN_DK;
                    ((unsigned short*)out)[(size_t)((b_ * ATTN_H + h_) * ATTN_DK + dk) * ATTN_S + s_] = f2bfbits(v);
                } else {
                    ((float*)out)[(size_t)m * N + n] = v;
                }
            }
        }
    }
}

// ------------------------- causal flash attention --------------------------
// Q,K: (B,S,H*DK) bf16 row-major.  Vt: (B,H,DK,S) bf16.  O: (B,S,H*DK) bf16.
// Grid: x = S/64 q-tiles, y = B*H.  Block = 128 thr = 4 waves, 16 q-rows/wave.
// K/V tiles staged once per block via async copies (shared by all 4 waves).

__global__ __launch_bounds__(128) void k_flash(const bf16* __restrict__ Q,
                                               const bf16* __restrict__ Kb,
                                               const bf16* __restrict__ Vt,
                                               unsigned short* __restrict__ O,
                                               float scale) {
    __shared__ __align__(32) bf16 ldsK[2][32 * 64];          // [buf][key*64+dk]
    __shared__ __align__(32) bf16 ldsV[2][64 * 32];          // [buf][dk*32+key]
    __shared__ __align__(32) unsigned short ldsP[4][16 * 32]; // per-wave P tile

    const int t    = threadIdx.x;
    const int lane = t & 31;
    const int wave = t >> 5;
    const int col = lane & 15, half = lane >> 4;

    const int bh = blockIdx.y;
    const int b = bh / ATTN_H, h = bh % ATTN_H;
    const int qw0 = blockIdx.x * 64 + wave * 16;

    const bf16* Qp = Q  + (size_t)b * ATTN_S * ATTN_D + h * ATTN_DK;
    const bf16* Kp = Kb + (size_t)b * ATTN_S * ATTN_D + h * ATTN_DK;
    const bf16* Vp = Vt + (size_t)(b * ATTN_H + h) * ATTN_DK * ATTN_S;
    unsigned short* Op = O + (size_t)b * ATTN_S * ATTN_D + h * ATTN_DK;

    // Issue K/V tile copies for 32 keys starting at kt0: 4 chunks/thread.
    auto issue_tile = [&](int kt0, int buf) {
        #pragma unroll
        for (int i = 0; i < 2; ++i) {            // K: 32 rows x 64 elems
            const int c = t + 128 * i;           // 256 chunks of 16B
            async_b128(Kp + (size_t)(kt0 + (c >> 3)) * ATTN_D + (c & 7) * 8,
                       &ldsK[buf][(c >> 3) * 64 + (c & 7) * 8]);
        }
        #pragma unroll
        for (int i = 0; i < 2; ++i) {            // V: 64 rows x 32 keys
            const int c = t + 128 * i;
            async_b128(Vp + (size_t)(c >> 2) * ATTN_S + kt0 + (c & 3) * 8,
                       &ldsV[buf][(c >> 2) * 32 + (c & 3) * 8]);
        }
    };

    // Q fragments for the whole key loop (contraction DK=64 -> 2x32).
    v16bf qf0 = load_a16x32(Qp + (size_t)qw0 * ATTN_D + 0,  ATTN_D);
    v16bf qf1 = load_a16x32(Qp + (size_t)qw0 * ATTN_D + 32, ATTN_D);

    v8f acc_o[4] = {};
    float mrow[8], lrow[8];
    #pragma unroll
    for (int r = 0; r < 8; ++r) { mrow[r] = -1e30f; lrow[r] = 0.0f; }

    const int niter = 2 * blockIdx.x + 2;        // causal bound (32 keys/iter)
    issue_tile(0, 0);
    for (int it = 0; it < niter; ++it) {
        const int buf = it & 1;
        const int more = (it + 1 < niter);
        if (more) issue_tile((it + 1) * 32, buf ^ 1);
        wait_async_issued(more, 4);
        __syncthreads();

        const int kt0 = it * 32;

        // ---- scores: S(16x32) = Q(16x64) @ K_tile^T ----
        v8f s[2];
        #pragma unroll
        for (int j = 0; j < 2; ++j) {
            v16bf kb0 = load_b32x16(&ldsK[buf][(16 * j) * 64 + 0],  64);
            v16bf kb1 = load_b32x16(&ldsK[buf][(16 * j) * 64 + 32], 64);
            v8f z = {};
            z = wmma_bf16(qf0, kb0, z);
            z = wmma_bf16(qf1, kb1, z);
            s[j] = z;
        }

        // ---- scale + causal mask ----
        #pragma unroll
        for (int j = 0; j < 2; ++j)
            #pragma unroll
            for (int r = 0; r < 8; ++r) {
                const int qrow = qw0 + r + 8 * half;
                const int kcol = kt0 + 16 * j + col;
                float sv = s[j][r] * scale;
                if (kcol > qrow) sv = -1e30f;
                s[j][r] = sv;
            }

        // ---- online softmax: row reductions over 16 lanes of each half ----
        float mnew[8];
        #pragma unroll
        for (int r = 0; r < 8; ++r)
            mnew[r] = fmaxf(mrow[r], fmaxf(s[0][r], s[1][r]));
        #pragma unroll
        for (int off = 1; off < 16; off <<= 1)
            #pragma unroll
            for (int r = 0; r < 8; ++r)
                mnew[r] = fmaxf(mnew[r], __shfl_xor(mnew[r], off, 32));

        float alpha[8], rs[8];
        #pragma unroll
        for (int r = 0; r < 8; ++r) {
            alpha[r] = __expf(mrow[r] - mnew[r]);
            mrow[r] = mnew[r];
            float p0 = __expf(s[0][r] - mnew[r]);
            float p1 = __expf(s[1][r] - mnew[r]);
            s[0][r] = p0; s[1][r] = p1;
            rs[r] = p0 + p1;
        }
        #pragma unroll
        for (int off = 1; off < 16; off <<= 1)
            #pragma unroll
            for (int r = 0; r < 8; ++r)
                rs[r] += __shfl_xor(rs[r], off, 32);
        #pragma unroll
        for (int r = 0; r < 8; ++r)
            lrow[r] = lrow[r] * alpha[r] + rs[r];

        #pragma unroll
        for (int jt = 0; jt < 4; ++jt)
            #pragma unroll
            for (int r = 0; r < 8; ++r)
                acc_o[jt][r] *= alpha[r];

        // ---- P (C layout) -> LDS -> A-operand layout ----
        unsigned short* lp = ldsP[wave];
        #pragma unroll
        for (int j = 0; j < 2; ++j)
            #pragma unroll
            for (int r = 0; r < 8; ++r)
                lp[(r + 8 * half) * 32 + 16 * j + col] = f2bfbits(s[j][r]);
        __syncthreads();                         // uniform (niter per-block)

        v16bf pf = load_a16x32((const bf16*)lp, 32);

        // ---- O += P(16x32) @ V_tile ----
        #pragma unroll
        for (int jt = 0; jt < 4; ++jt) {
            v16bf vb = load_b32x16(&ldsV[buf][(16 * jt) * 32], 32);
            acc_o[jt] = wmma_bf16(pf, vb, acc_o[jt]);
        }
        __syncthreads();                         // protect ldsP + K/V buffers
    }

    // ---- normalize and store ----
    #pragma unroll
    for (int jt = 0; jt < 4; ++jt)
        #pragma unroll
        for (int r = 0; r < 8; ++r) {
            const int q = qw0 + r + 8 * half;
            const float o = acc_o[jt][r] / lrow[r];
            Op[(size_t)q * ATTN_D + 16 * jt + col] = f2bfbits(o);
        }
}

// ------------------------------- launcher ----------------------------------

extern "C" void kernel_launch(void* const* d_in, const int* in_sizes, int n_in,
                              void* d_out, int out_size, void* d_ws, size_t ws_size,
                              hipStream_t stream) {
    (void)in_sizes; (void)n_in; (void)out_size; (void)ws_size;

    const float* x  = (const float*)d_in[0];
    const float* wq = (const float*)d_in[1];
    const float* wk = (const float*)d_in[2];
    const float* wv = (const float*)d_in[3];
    const float* wo = (const float*)d_in[4];

    const int M = ATTN_M, D = ATTN_D;
    const size_t nx = (size_t)M * D;        // 4,194,304
    const size_t nw = (size_t)D * D;        // 1,048,576

    // Carve bf16 workspace (256B aligned).  Total ~48 MB.
    size_t off = 0;
    auto carve = [&](size_t bytes) {
        void* p = (char*)d_ws + off;
        off += (bytes + 255) & ~(size_t)255;
        return p;
    };
    bf16* x_bf  = (bf16*)carve(nx * 2);
    bf16* wq_bf = (bf16*)carve(nw * 2);
    bf16* wk_bf = (bf16*)carve(nw * 2);
    bf16* wv_bf = (bf16*)carve(nw * 2);
    bf16* wo_bf = (bf16*)carve(nw * 2);
    bf16* q_bf  = (bf16*)carve(nx * 2);
    bf16* k_bf  = (bf16*)carve(nx * 2);
    bf16* vt_bf = (bf16*)carve(nx * 2);     // (B,H,DK,S)
    bf16* o_bf  = (bf16*)carve(nx * 2);

    // 1) fp32 -> bf16 conversions
    k_cvt_f32_bf16<<<dim3((unsigned)(nx / 1024)), dim3(256), 0, stream>>>(x,  (unsigned short*)x_bf,  (int)nx);
    k_cvt_f32_bf16<<<dim3((unsigned)(nw / 1024)), dim3(256), 0, stream>>>(wq, (unsigned short*)wq_bf, (int)nw);
    k_cvt_f32_bf16<<<dim3((unsigned)(nw / 1024)), dim3(256), 0, stream>>>(wk, (unsigned short*)wk_bf, (int)nw);
    k_cvt_f32_bf16<<<dim3((unsigned)(nw / 1024)), dim3(256), 0, stream>>>(wv, (unsigned short*)wv_bf, (int)nw);
    k_cvt_f32_bf16<<<dim3((unsigned)(nw / 1024)), dim3(256), 0, stream>>>(wo, (unsigned short*)wo_bf, (int)nw);

    // 2) Q/K/V projections (async-staged WMMA GEMMs)
    dim3 ggrid(M / 256, D / 64), gblk(256);
    k_gemm_xwt<<<ggrid, gblk, 0, stream>>>(x_bf, wq_bf, q_bf,  M, D, D, 0);
    k_gemm_xwt<<<ggrid, gblk, 0, stream>>>(x_bf, wk_bf, k_bf,  M, D, D, 0);
    k_gemm_xwt<<<ggrid, gblk, 0, stream>>>(x_bf, wv_bf, vt_bf, M, D, D, 1);

    // 3) causal flash attention (async-staged K/V tiles)
    k_flash<<<dim3(ATTN_S / 64, ATTN_B * ATTN_H), dim3(128), 0, stream>>>(
        q_bf, k_bf, vt_bf, (unsigned short*)o_bf, 0.125f /* 1/sqrt(64) */);

    // 4) output projection -> f32 d_out
    k_gemm_xwt<<<ggrid, gblk, 0, stream>>>(o_bf, wo_bf, d_out, M, D, D, 2);
}
